// SwinTransformerBlock_44848048505306
// MI455X (gfx1250) — matrix operand order repaired and checked by reading, not compiled, not run on hardware
//
#include <hip/hip_runtime.h>
#include <hip/hip_fp16.h>

typedef _Float16 h16;
typedef __attribute__((ext_vector_type(8)))  _Float16 v8h;
typedef __attribute__((ext_vector_type(16))) _Float16 v16h;
typedef __attribute__((ext_vector_type(8)))  float    v8f;
typedef __attribute__((ext_vector_type(4)))  int      v4i;

#define WMMA_F32_F16(a,b,c) \
  __builtin_amdgcn_wmma_f32_16x16x32_f16(false,(a),false,(b),(short)0,(c),false,false)

// ---- async global->LDS copy (CDNA5), with synchronous fallback ----
#if defined(__has_builtin)
#  if __has_builtin(__builtin_amdgcn_global_load_async_to_lds_b128) && \
      __has_builtin(__builtin_amdgcn_s_wait_asynccnt)
#    define HAVE_ASYNC 1
#  endif
#endif
#ifndef HAVE_ASYNC
#  define HAVE_ASYNC 0
#endif

__device__ __forceinline__ void cp_b128(const h16* g, h16* l) {
#if HAVE_ASYNC
  __builtin_amdgcn_global_load_async_to_lds_b128(
      (__attribute__((address_space(1))) v4i*)(void*)g,
      (__attribute__((address_space(3))) v4i*)(void*)l, 0, 0);
#else
  *(uint4*)l = *(const uint4*)g;
#endif
}
__device__ __forceinline__ void wait_async() {
#if HAVE_ASYNC
  __builtin_amdgcn_s_wait_asynccnt(0);
#endif
}

// ---- problem constants ----
// B=32 H=W=56 C=192 NH=6 WS=7 SS=3 N=49 HD=32 NW=64 HID=768
static constexpr int MTOK = 32 * 64 * 49;   // 100352 tokens (window order)
static constexpr int CDIM = 192;

__device__ __forceinline__ v16h cat8(v8h a, v8h b) {
  return __builtin_shufflevector(a, b, 0,1,2,3,4,5,6,7,8,9,10,11,12,13,14,15);
}

// window-order row m  <->  (b, h, w) token index in the original image
__device__ __forceinline__ int map_token(int m) {
  int win = m / 49, n = m % 49;
  int b  = win >> 6, wi = win & 63;
  int wh = wi >> 3,  ww = wi & 7;
  int r  = n / 7,    c  = n % 7;
  int hs = wh * 7 + r + 3; if (hs >= 56) hs -= 56;
  int ws = ww * 7 + c + 3; if (ws >= 56) ws -= 56;
  return b * 3136 + hs * 56 + ws;
}

// ---------------- prep: weight transpose to f16 (Wt[n][k]) ----------------
__global__ __launch_bounds__(256)
void k_transpose_w(const float* __restrict__ W, h16* __restrict__ Wt, int K, int N) {
  int idx = blockIdx.x * 256 + threadIdx.x;
  if (idx >= K * N) return;
  int k = idx / N, n = idx % N;
  Wt[(size_t)n * K + k] = (h16)W[idx];
}

// ---------------- prep: relative position bias table (NH,49,49) -----------
__global__ __launch_bounds__(256)
void k_bias_attn(const float* __restrict__ rpb, float* __restrict__ battn) {
  int idx = blockIdx.x * 256 + threadIdx.x;
  if (idx >= 6 * 49 * 49) return;
  int h = idx / 2401, rem = idx % 2401;
  int n = rem / 49, m = rem % 49;
  int rn = n / 7, cn = n % 7, rm = m / 7, cm = m % 7;
  int ridx = (rn - rm + 6) * 13 + (cn - cm + 6);
  battn[idx] = rpb[ridx * 6 + h];
}

// ---------------- LayerNorm (optionally fused shift+window-partition) -----
__global__ __launch_bounds__(256)
void k_layernorm(const float* __restrict__ src, const float* __restrict__ g,
                 const float* __restrict__ bt, h16* __restrict__ dst, int mapped) {
  int wid = threadIdx.x >> 5, lane = threadIdx.x & 31;
  int m = blockIdx.x * 8 + wid;                  // 12544*8 == 100352 exactly
  int srow = mapped ? map_token(m) : m;
  const float* p = src + (size_t)srow * CDIM;
  float v[6], s = 0.f;
#pragma unroll
  for (int j = 0; j < 6; ++j) { v[j] = p[lane + 32 * j]; s += v[j]; }
#pragma unroll
  for (int o = 16; o >= 1; o >>= 1) s += __shfl_xor(s, o, 32);
  float mu = s * (1.f / 192.f), vs = 0.f;
#pragma unroll
  for (int j = 0; j < 6; ++j) { float d = v[j] - mu; vs += d * d; }
#pragma unroll
  for (int o = 16; o >= 1; o >>= 1) vs += __shfl_xor(vs, o, 32);
  float rs = rsqrtf(vs * (1.f / 192.f) + 1e-5f);
#pragma unroll
  for (int j = 0; j < 6; ++j) {
    int col = lane + 32 * j;
    dst[(size_t)m * CDIM + col] = (h16)((v[j] - mu) * rs * g[col] + bt[col]);
  }
}

// ---------------- generic WMMA GEMM: Y = A(M,K) * Wt(N,K)^T + bias --------
// Double-buffered LDS tiles filled with async global->LDS copies.
// mode 0: store f16    mode 1: GELU, store f16
// mode 2: proj epilogue: d_out[T(m)] = resid[T(m)] + val (window-reverse scatter)
// mode 3: d_out[m] += val
__global__ __launch_bounds__(256)
void k_gemm(const h16* __restrict__ A, const h16* __restrict__ Wt,
            const float* __restrict__ bias, int K, int Ncols, int mode,
            h16* __restrict__ out16, float* __restrict__ outf,
            const float* __restrict__ resid) {
  __shared__ h16 As[2][128 * 40];
  __shared__ h16 Bs[2][64 * 40];
  const int tid = threadIdx.x, wid = tid >> 5, lane = tid & 31;
  const int hi = lane >> 4, l15 = lane & 15;
  const int m0 = blockIdx.x * 128, n0 = blockIdx.y * 64;
  const int r0 = tid >> 2, kq = tid & 3;         // per-thread staging coords

  v8f acc[4];
#pragma unroll
  for (int t = 0; t < 4; ++t) acc[t] = v8f{0.f,0.f,0.f,0.f,0.f,0.f,0.f,0.f};

  const h16* gA = A  + (size_t)(m0 + r0) * K + kq * 8;
  const h16* gB = Wt + (size_t)(n0 + r0) * K + kq * 8;

  auto issue = [&](int buf, int k0) {
    cp_b128(gA + k0,                   &As[buf][r0 * 40 + kq * 8]);
    cp_b128(gA + (size_t)64 * K + k0,  &As[buf][(r0 + 64) * 40 + kq * 8]);
    cp_b128(gB + k0,                   &Bs[buf][r0 * 40 + kq * 8]);
  };

  const int nk = K >> 5;
  issue(0, 0);
  for (int i = 0; i < nk; ++i) {
    wait_async();                       // my tile-i pieces have landed in LDS
    __syncthreads();                    // everyone's pieces landed; prev buffer fully consumed
    if (i + 1 < nk) issue((i + 1) & 1, (i + 1) * 32);
    const h16* as = As[i & 1];
    const h16* bs = Bs[i & 1];
    int arow = wid * 16 + l15;
    v16h a = cat8(*(const v8h*)&as[arow * 40 + hi * 8],
                  *(const v8h*)&as[arow * 40 + 16 + hi * 8]);
#pragma unroll
    for (int t = 0; t < 4; ++t) {
      int brow = t * 16 + l15;
      v8h blo = *(const v8h*)&bs[brow * 40 + hi * 8];
      v8h bhi = *(const v8h*)&bs[brow * 40 + 16 + hi * 8];
      acc[t] = WMMA_F32_F16(a, cat8(blo, bhi), acc[t]);
    }
  }

  int tok[8];
  if (mode == 2) {
#pragma unroll
    for (int r = 0; r < 8; ++r) tok[r] = map_token(m0 + wid * 16 + r + 8 * hi);
  }
#pragma unroll
  for (int t = 0; t < 4; ++t) {
    int col = n0 + t * 16 + l15;
    float bv = bias[col];
#pragma unroll
    for (int r = 0; r < 8; ++r) {
      int m = m0 + wid * 16 + r + 8 * hi;
      float val = acc[t][r] + bv;
      if (mode == 1) val = 0.5f * val * (1.f + erff(val * 0.70710678f));
      if (mode <= 1) {
        out16[(size_t)m * Ncols + col] = (h16)val;
      } else if (mode == 2) {
        size_t d = (size_t)tok[r] * CDIM + col;
        outf[d] = resid[d] + val;
      } else {
        outf[(size_t)m * CDIM + col] += val;
      }
    }
  }
}

// ---------------- windowed attention: one block per (window, head) --------
__global__ __launch_bounds__(128)
void k_attn(const h16* __restrict__ qkv, const float* __restrict__ battn,
            const float* __restrict__ mask, h16* __restrict__ o16) {
  __shared__ h16 Qs[64 * 40];
  __shared__ h16 Ks[64 * 40];
  __shared__ h16 Vt[32 * 72];
  __shared__ h16 Ps[64 * 72];
  int bid = blockIdx.x;
  int win = bid / 6, hd = bid % 6;
  int wi = win & 63;
  int tid = threadIdx.x, wid = tid >> 5, lane = tid & 31;
  int hi = lane >> 4, l15 = lane & 15;
  size_t base = (size_t)win * 49 * 576;
  int qo = hd * 32, ko = 192 + hd * 32, vo = 384 + hd * 32;

  for (int q = tid; q < 256; q += 128) {
    int row = q >> 2, kq = q & 3;
    if (row < 49) {
      const h16* src = qkv + base + (size_t)row * 576;
      cp_b128(src + qo + kq * 8, &Qs[row * 40 + kq * 8]);   // async Q tile
      cp_b128(src + ko + kq * 8, &Ks[row * 40 + kq * 8]);   // async K tile
      v8h vv = *(const v8h*)(src + vo + kq * 8);            // V: transpose via VGPR
#pragma unroll
      for (int j = 0; j < 8; ++j) Vt[(kq * 8 + j) * 72 + row] = vv[j];
    } else {
      uint4 z = make_uint4(0, 0, 0, 0);
      *(uint4*)&Qs[row * 40 + kq * 8] = z;
      *(uint4*)&Ks[row * 40 + kq * 8] = z;
#pragma unroll
      for (int j = 0; j < 8; ++j) Vt[(kq * 8 + j) * 72 + row] = (h16)0.f;
    }
  }
  wait_async();
  __syncthreads();

  // S = Q * K^T  (one wmma per 16x16 tile, K=HD=32)
  int arow = wid * 16 + l15;
  v16h a = cat8(*(const v8h*)&Qs[arow * 40 + hi * 8],
                *(const v8h*)&Qs[arow * 40 + 16 + hi * 8]);
  v8f s[4];
#pragma unroll
  for (int t = 0; t < 4; ++t) {
    s[t] = v8f{0.f,0.f,0.f,0.f,0.f,0.f,0.f,0.f};
    int key = t * 16 + l15;
    v16h b = cat8(*(const v8h*)&Ks[key * 40 + hi * 8],
                  *(const v8h*)&Ks[key * 40 + 16 + hi * 8]);
    s[t] = WMMA_F32_F16(a, b, s[t]);
  }

  // logits + bias + mask, row-wise softmax in registers (16-lane groups)
  const float scale = 0.1767766953f;   // HD^-0.5
  float l[4][8];
#pragma unroll
  for (int t = 0; t < 4; ++t) {
    int col = t * 16 + l15;
#pragma unroll
    for (int r = 0; r < 8; ++r) {
      int row = wid * 16 + r + 8 * hi;
      float val = -1e30f;
      if (row < 49 && col < 49)
        val = s[t][r] * scale + battn[hd * 2401 + row * 49 + col]
                              + mask[(size_t)wi * 2401 + row * 49 + col];
      l[t][r] = val;
    }
  }
#pragma unroll
  for (int r = 0; r < 8; ++r) {
    float mx = fmaxf(fmaxf(l[0][r], l[1][r]), fmaxf(l[2][r], l[3][r]));
#pragma unroll
    for (int o = 8; o >= 1; o >>= 1) mx = fmaxf(mx, __shfl_xor(mx, o, 16));
    float sm = 0.f;
#pragma unroll
    for (int t = 0; t < 4; ++t) { l[t][r] = __expf(l[t][r] - mx); sm += l[t][r]; }
#pragma unroll
    for (int o = 8; o >= 1; o >>= 1) sm += __shfl_xor(sm, o, 16);
    float inv = 1.f / sm;
    int row = wid * 16 + r + 8 * hi;
#pragma unroll
    for (int t = 0; t < 4; ++t)
      Ps[row * 72 + t * 16 + l15] = (h16)(l[t][r] * inv);
  }
  __syncthreads();

  // O = P * V   (64x64 @ 64x32, two K-chunks of 32)
  v8f o[2];
#pragma unroll
  for (int t = 0; t < 2; ++t) o[t] = v8f{0.f,0.f,0.f,0.f,0.f,0.f,0.f,0.f};
#pragma unroll
  for (int kc = 0; kc < 2; ++kc) {
    v16h ap = cat8(*(const v8h*)&Ps[arow * 72 + kc * 32 + hi * 8],
                   *(const v8h*)&Ps[arow * 72 + kc * 32 + 16 + hi * 8]);
#pragma unroll
    for (int t = 0; t < 2; ++t) {
      int d = t * 16 + l15;
      v16h bv = cat8(*(const v8h*)&Vt[d * 72 + kc * 32 + hi * 8],
                     *(const v8h*)&Vt[d * 72 + kc * 32 + 16 + hi * 8]);
      o[t] = WMMA_F32_F16(ap, bv, o[t]);
    }
  }
#pragma unroll
  for (int t = 0; t < 2; ++t) {
    int col = hd * 32 + t * 16 + l15;
#pragma unroll
    for (int r = 0; r < 8; ++r) {
      int row = wid * 16 + r + 8 * hi;
      if (row < 49)
        o16[((size_t)win * 49 + row) * CDIM + col] = (h16)o[t][r];
    }
  }
}

// -------------------------------- launcher --------------------------------
extern "C" void kernel_launch(void* const* d_in, const int* in_sizes, int n_in,
                              void* d_out, int out_size, void* d_ws, size_t ws_size,
                              hipStream_t stream) {
  const float* x      = (const float*)d_in[0];
  const float* amask  = (const float*)d_in[1];
  const float* ln1_g  = (const float*)d_in[2];
  const float* ln1_b  = (const float*)d_in[3];
  const float* qkv_w  = (const float*)d_in[4];
  const float* qkv_b  = (const float*)d_in[5];
  const float* rpb    = (const float*)d_in[6];
  const float* proj_w = (const float*)d_in[7];
  const float* proj_b = (const float*)d_in[8];
  const float* ln2_g  = (const float*)d_in[9];
  const float* ln2_b  = (const float*)d_in[10];
  const float* fc1_w  = (const float*)d_in[11];
  const float* fc1_b  = (const float*)d_in[12];
  const float* fc2_w  = (const float*)d_in[13];
  const float* fc2_b  = (const float*)d_in[14];
  float* out = (float*)d_out;

  char* ws = (char*)d_ws;
  size_t off = 0;
  h16* h16buf = (h16*)(ws + off); off += (size_t)MTOK * 192 * 2;   // LN1 / LN2 out
  h16* qkv16  = (h16*)(ws + off); off += (size_t)MTOK * 576 * 2;
  h16* o16    = (h16*)(ws + off); off += (size_t)MTOK * 192 * 2;
  h16* m16    = (h16*)(ws + off); off += (size_t)MTOK * 768 * 2;   // MLP hidden
  h16* wt_qkv = (h16*)(ws + off); off += (size_t)576 * 192 * 2;
  h16* wt_prj = (h16*)(ws + off); off += (size_t)192 * 192 * 2;
  h16* wt_fc1 = (h16*)(ws + off); off += (size_t)768 * 192 * 2;
  h16* wt_fc2 = (h16*)(ws + off); off += (size_t)192 * 768 * 2;
  float* battn = (float*)(ws + off); off += (size_t)6 * 49 * 49 * 4;

  // prep
  k_transpose_w<<<(192*576 + 255)/256, 256, 0, stream>>>(qkv_w,  wt_qkv, 192, 576);
  k_transpose_w<<<(192*192 + 255)/256, 256, 0, stream>>>(proj_w, wt_prj, 192, 192);
  k_transpose_w<<<(192*768 + 255)/256, 256, 0, stream>>>(fc1_w,  wt_fc1, 192, 768);
  k_transpose_w<<<(768*192 + 255)/256, 256, 0, stream>>>(fc2_w,  wt_fc2, 768, 192);
  k_bias_attn<<<(6*49*49 + 255)/256, 256, 0, stream>>>(rpb, battn);

  // LN1 + shift + window partition -> f16
  k_layernorm<<<MTOK/8, 256, 0, stream>>>(x, ln1_g, ln1_b, h16buf, 1);
  // QKV projection
  k_gemm<<<dim3(MTOK/128, 576/64), 256, 0, stream>>>(
      h16buf, wt_qkv, qkv_b, 192, 576, 0, qkv16, nullptr, nullptr);
  // attention per (window, head)
  k_attn<<<2048 * 6, 128, 0, stream>>>(qkv16, battn, amask, o16);
  // proj + window-reverse + unshift + residual -> x1 in d_out
  k_gemm<<<dim3(MTOK/128, 192/64), 256, 0, stream>>>(
      o16, wt_prj, proj_b, 192, 192, 2, nullptr, out, x);
  // LN2
  k_layernorm<<<MTOK/8, 256, 0, stream>>>(out, ln2_g, ln2_b, h16buf, 0);
  // fc1 + GELU
  k_gemm<<<dim3(MTOK/128, 768/64), 256, 0, stream>>>(
      h16buf, wt_fc1, fc1_b, 192, 768, 1, m16, nullptr, nullptr);
  // fc2 + residual accumulate into d_out
  k_gemm<<<dim3(MTOK/128, 192/64), 256, 0, stream>>>(
      m16, wt_fc2, fc2_b, 768, 192, 3, nullptr, out, nullptr);
}
